// Y_prime_decoder_5583457485495
// MI455X (gfx1250) — compile-verified, matrix operand order checked by reading, not run on hardware
//
#include <hip/hip_runtime.h>

#define N_NODES 50000
#define N_EDGES 800000
#define NFEAT   128
#define NHID    512
#define NOUT    2

typedef __attribute__((ext_vector_type(16))) __bf16          v16bf;
typedef __attribute__((ext_vector_type(16))) unsigned short  v16u;
typedef __attribute__((ext_vector_type(8)))  unsigned int    v8u;
typedef __attribute__((ext_vector_type(8)))  float           v8f;

union BfVec { v16bf bf; v16u u16; v8u u32; };

// pack two f32 -> packed bf16x2 (round-to-nearest w/ bias), one v_perm_b32
__device__ __forceinline__ unsigned int pack_bf16(float lo, float hi) {
  unsigned int a = __float_as_uint(lo) + 0x8000u;
  unsigned int b = __float_as_uint(hi) + 0x8000u;
  // result bytes [3:0] = { b[3], b[2], a[3], a[2] }  (src0=b -> idx 4..7, src1=a -> idx 0..3)
  return __builtin_amdgcn_perm(b, a, 0x07060302u);
}
__device__ __forceinline__ unsigned short f2bf(float f) {
  return (unsigned short)((__float_as_uint(f) + 0x8000u) >> 16);
}
__device__ __forceinline__ float bf2f(unsigned short h) {
  return __uint_as_float(((unsigned int)h) << 16);
}

// ---------------------------------------------------------------- utilities
__global__ void zero_kernel(float4* __restrict__ p, int n4) {
  int i = blockIdx.x * blockDim.x + threadIdx.x;
  if (i < n4) p[i] = make_float4(0.f, 0.f, 0.f, 0.f);
}

__global__ void deg_kernel(const long long* __restrict__ dst,
                           float* __restrict__ deg) {
  int e = blockIdx.x * blockDim.x + threadIdx.x;
  if (e < N_EDGES) atomicAdd(deg + (int)dst[e], 1.0f);
}

__global__ void dinv_kernel(float* __restrict__ deg) {
  int i = blockIdx.x * blockDim.x + threadIdx.x;
  if (i < N_NODES) deg[i] = rsqrtf(deg[i] + 1.0f);  // deg includes self-loop
}

// ------------------------------------------- pre-pack W1 into WMMA B layout
// One thread per (tile = ntile*4+kt, lane): emits 8 dwords = 16 bf16 = this
// lane's B operand for that 16-col x 32-K chunk. 128 tiles x 32 lanes.
__global__ void pack_w1(const float* __restrict__ W1,
                        unsigned int* __restrict__ pw1) {
  const int gid  = blockIdx.x * blockDim.x + threadIdx.x;  // 0..4095
  const int t    = gid >> 5;                               // 0..127
  const int lane = gid & 31;
  if (t >= (NHID / 16) * 4) return;
  const int ntile = t >> 2;
  const int kt    = t & 3;
  const int col   = ntile * 16 + (lane & 15);
  const int khalf = (lane >> 4) * 8;
  unsigned int* out = pw1 + ((size_t)t * 32 + lane) * 8;
#pragma unroll
  for (int v = 0; v < 8; ++v) {
    const int k0 = kt * 32 + khalf + ((v < 4) ? (2 * v) : (16 + 2 * (v - 4)));
    out[v] = pack_bf16(W1[(size_t)k0 * NHID + col],
                       W1[(size_t)(k0 + 1) * NHID + col]);
  }
}

// ------------------------------------------------- layer-1 edge aggregation
// aggX[dst] += dinv[src]*dinv[dst] * X[src]  (128 channels, 1 wave per edge)
__global__ void edge_agg1(const long long* __restrict__ src,
                          const long long* __restrict__ dst,
                          const float* __restrict__ dinv,
                          const float* __restrict__ X,
                          float* __restrict__ aggX) {
  const int gid  = blockIdx.x * blockDim.x + threadIdx.x;
  const int e    = gid >> 5;
  const int lane = gid & 31;
  if (e >= N_EDGES) return;
  const int s = (int)src[e];
  const int d = (int)dst[e];
  const float w = dinv[s] * dinv[d];
  const float4 xv = *reinterpret_cast<const float4*>(X + (size_t)s * NFEAT + lane * 4);
  float* out = aggX + (size_t)d * NFEAT + lane * 4;
  atomicAdd(out + 0, w * xv.x);
  atomicAdd(out + 1, w * xv.y);
  atomicAdd(out + 2, w * xv.z);
  atomicAdd(out + 3, w * xv.w);
}

// ---------------------------------------------------------- layer-1 matmul
// H1 = relu( (aggX + dinv^2 * X) @ W1 + b1 )  via v_wmma_f32_16x16x32_bf16
// One wave computes a 16x16 tile; K = 128 = 4 WMMA of K=32.
__global__ void gemm1_wmma(const float* __restrict__ aggX,
                           const float* __restrict__ X,
                           const float* __restrict__ dinv,
                           const unsigned int* __restrict__ pw1,
                           const float* __restrict__ b1,
                           unsigned short* __restrict__ H1) {
  const int lane  = threadIdx.x & 31;
  const int wave  = threadIdx.x >> 5;
  const int mtile = blockIdx.x;              // 0..3124
  const int ntile = blockIdx.y * 4 + wave;   // 0..31
  const int row   = mtile * 16 + (lane & 15);
  const int col   = ntile * 16 + (lane & 15);
  const int khalf = (lane >> 4) * 8;

  const float dv = dinv[row];
  const float d2 = dv * dv;
  const float* ax = aggX + (size_t)row * NFEAT;
  const float* xr = X    + (size_t)row * NFEAT;

  v8f acc = {};
#pragma unroll
  for (int kt = 0; kt < 4; ++kt) {
    const int kbase = kt * 32;
    BfVec a, b;
    // B: pre-packed, lane-contiguous -> two coalesced b128 loads
    const uint4* bp = reinterpret_cast<const uint4*>(
        pw1 + ((size_t)(ntile * 4 + kt) * 32 + lane) * 8);
    const uint4 bv0 = bp[0];
    const uint4 bv1 = bp[1];
    b.u32[0] = bv0.x; b.u32[1] = bv0.y; b.u32[2] = bv0.z; b.u32[3] = bv0.w;
    b.u32[4] = bv1.x; b.u32[5] = bv1.y; b.u32[6] = bv1.z; b.u32[7] = bv1.w;
    // A: fused self-loop add, pack pairs with v_perm
#pragma unroll
    for (int v = 0; v < 8; ++v) {
      const int k0 = kbase + khalf + ((v < 4) ? (2 * v) : (16 + 2 * (v - 4)));
      a.u32[v] = pack_bf16(ax[k0]     + d2 * xr[k0],
                           ax[k0 + 1] + d2 * xr[k0 + 1]);
    }
    acc = __builtin_amdgcn_wmma_f32_16x16x32_bf16(false, a.bf, false, b.bf,
                                                  (short)0, acc, false, false);
  }

  const float bias = b1[col];
  const int mofs = (lane >> 4) * 8;   // C/D layout: VGPR r holds M = mofs + r
#pragma unroll
  for (int r = 0; r < 8; ++r) {
    const int m = mtile * 16 + mofs + r;
    float val = fmaxf(acc[r] + bias, 0.0f);
    H1[(size_t)m * NHID + col] = f2bf(val);
  }
}

// ------------------------------------------------- layer-2 transform (512->2)
// xw2[node] = H1[node] @ W2 ; one wave per node, shuffle reduction
__global__ void gemm2_dot(const unsigned short* __restrict__ H1,
                          const float* __restrict__ W2,
                          float* __restrict__ xw2) {
  const int gid  = blockIdx.x * blockDim.x + threadIdx.x;
  const int node = gid >> 5;
  const int lane = gid & 31;
  if (node >= N_NODES) return;
  const unsigned short* h = H1 + (size_t)node * NHID;
  float a0 = 0.f, a1 = 0.f;
#pragma unroll 4
  for (int k = lane; k < NHID; k += 32) {
    float hv = bf2f(h[k]);
    a0 += hv * W2[2 * k];
    a1 += hv * W2[2 * k + 1];
  }
#pragma unroll
  for (int off = 16; off > 0; off >>= 1) {
    a0 += __shfl_down(a0, off, 32);
    a1 += __shfl_down(a1, off, 32);
  }
  if (lane == 0) {
    xw2[2 * (size_t)node]     = a0;
    xw2[2 * (size_t)node + 1] = a1;
  }
}

// ------------------------------------------------- layer-2 edge aggregation
__global__ void edge_agg2(const long long* __restrict__ src,
                          const long long* __restrict__ dst,
                          const float* __restrict__ dinv,
                          const float* __restrict__ xw2,
                          float* __restrict__ agg2) {
  int e = blockIdx.x * blockDim.x + threadIdx.x;
  if (e >= N_EDGES) return;
  int s = (int)src[e];
  int d = (int)dst[e];
  float w = dinv[s] * dinv[d];
  float2 v = *reinterpret_cast<const float2*>(xw2 + 2 * (size_t)s);
  atomicAdd(agg2 + 2 * (size_t)d,     w * v.x);
  atomicAdd(agg2 + 2 * (size_t)d + 1, w * v.y);
}

// ---------------------------------- self-loop + bias + softmax (2 channels)
__global__ void final_softmax(const float* __restrict__ agg2,
                              const float* __restrict__ xw2,
                              const float* __restrict__ dinv,
                              const float* __restrict__ b2,
                              float* __restrict__ out) {
  int i = blockIdx.x * blockDim.x + threadIdx.x;
  if (i >= N_NODES) return;
  float dv = dinv[i];
  float d2 = dv * dv;
  float l0 = agg2[2 * (size_t)i]     + d2 * xw2[2 * (size_t)i]     + b2[0];
  float l1 = agg2[2 * (size_t)i + 1] + d2 * xw2[2 * (size_t)i + 1] + b2[1];
  float m  = fmaxf(l0, l1);
  float e0 = __expf(l0 - m);
  float e1 = __expf(l1 - m);
  float inv = 1.0f / (e0 + e1);
  out[2 * (size_t)i]     = e0 * inv;
  out[2 * (size_t)i + 1] = e1 * inv;
}

extern "C" void kernel_launch(void* const* d_in, const int* in_sizes, int n_in,
                              void* d_out, int out_size, void* d_ws, size_t ws_size,
                              hipStream_t stream) {
  const float*     X     = (const float*)d_in[0];
  const long long* edges = (const long long*)d_in[1];   // int64 in reference
  const float*     W1    = (const float*)d_in[2];
  const float*     b1    = (const float*)d_in[3];
  const float*     W2    = (const float*)d_in[4];
  const float*     b2    = (const float*)d_in[5];
  const long long* src = edges;
  const long long* dst = edges + N_EDGES;

  // workspace layout (bytes):
  //   [0, 200000)              deg -> dinv (in place)
  //   [200000, 600000)         agg2   (50000*2 f32)
  //   [600000, 26200000)       aggX   (50000*128 f32)
  //   [26200000, 26600000)     xw2    (50000*2 f32)
  //   [26600000, 77800000)     H1     (50000*512 bf16)
  //   [77800000, 77931072)     pw1    (packed bf16 W1, WMMA B layout)
  char* ws = (char*)d_ws;
  float*          deg  = (float*)(ws);
  float*          agg2 = (float*)(ws + 200000);
  float*          aggX = (float*)(ws + 600000);
  float*          xw2  = (float*)(ws + 26200000);
  unsigned short* H1   = (unsigned short*)(ws + 26600000);
  unsigned int*   pw1  = (unsigned int*)(ws + 77800000);

  // 1. zero [deg | agg2 | aggX] = first 26,200,000 bytes
  const int n4 = 26200000 / 16;
  zero_kernel<<<(n4 + 255) / 256, 256, 0, stream>>>((float4*)ws, n4);

  // pre-pack W1 (tiny: 4096 threads)
  pack_w1<<<16, 256, 0, stream>>>(W1, pw1);

  // 2-3. degree + dinv
  deg_kernel<<<(N_EDGES + 255) / 256, 256, 0, stream>>>(dst, deg);
  dinv_kernel<<<(N_NODES + 255) / 256, 256, 0, stream>>>(deg);

  // 4. aggregate raw X over edges (1 wave/edge, 4 channels/lane)
  edge_agg1<<<(N_EDGES * 32) / 256, 256, 0, stream>>>(src, dst, deg, X, aggX);

  // 5. WMMA GEMM + self-loop + bias + relu -> bf16 hidden
  dim3 g1(N_NODES / 16, NHID / 64);   // (3125, 8) blocks of 4 waves
  gemm1_wmma<<<g1, 128, 0, stream>>>(aggX, X, deg, pw1, b1, H1);

  // 6. hidden -> 2 logits per node
  gemm2_dot<<<(N_NODES * 32 + 255) / 256, 256, 0, stream>>>(H1, W2, xw2);

  // 7. aggregate 2-channel logits over edges
  edge_agg2<<<(N_EDGES + 255) / 256, 256, 0, stream>>>(src, dst, deg, xw2, agg2);

  // 8. self-loop + bias + softmax
  final_softmax<<<(N_NODES + 255) / 256, 256, 0, stream>>>(agg2, xw2, deg, b2,
                                                           (float*)d_out);
}
